// Noise_VPT_13211319403315
// MI455X (gfx1250) — matrix operand order repaired and checked by reading, not compiled, not run on hardware
//
#include <hip/hip_runtime.h>
#include <math.h>

// ---- CDNA5 WMMA / vector types ----
typedef __attribute__((ext_vector_type(16))) __bf16          v16bf;
typedef __attribute__((ext_vector_type(16))) unsigned short  v16u;
typedef __attribute__((ext_vector_type(8)))  float           v8f;
typedef __attribute__((ext_vector_type(4)))  float           f32x4;
typedef __attribute__((ext_vector_type(4)))  unsigned short  u16x4;

#define C_DIM   512          // embedding dim (K)
#define P_DIM   1024         // number of centroids
#define KC      32           // K per bf16 WMMA
#define M_BLK   64           // rows per workgroup
#define NG      64           // columns per group iteration
#define NTILES  4            // 16-col WMMA tiles per group
#define NGROUPS (P_DIM / NG) // 16
#define KCHUNKS (C_DIM / KC) // 16
#define NCHUNKS (NGROUPS * KCHUNKS) // 256 pipelined chunks
#define THREADS 128
#define WAVES   4
#define LDS_STR 40           // padded row stride -> conflict-free b128 frag loads
#define ABUF    (M_BLK * LDS_STR)
#define BBUF    (NG * LDS_STR)

// f32 -> bf16 (RNE) as raw ushort, and back.
static __device__ inline unsigned short f2bf(float x) {
    unsigned u = __builtin_bit_cast(unsigned, x);
    unsigned r = (u + 0x7FFFu + ((u >> 16) & 1u)) >> 16;
    return (unsigned short)r;
}
static __device__ inline float bf2f(unsigned short s) {
    unsigned u = ((unsigned)s) << 16;
    return __builtin_bit_cast(float, u);
}

// insert d into ascending (t0<=t1<=t2) keeping the 3 smallest
#define TOP3_INSERT(d, t0, t1, t2)                  \
    {                                               \
        float _a1 = fmaxf((d), (t0));               \
        (t0) = fminf((d), (t0));                    \
        float _b2 = fmaxf(_a1, (t1));               \
        (t1) = fminf(_a1, (t1));                    \
        (t2) = fminf(_b2, (t2));                    \
    }

// Stage chunk q_ (group q_>>4, k-chunk q_&15) into LDS buffer b_.
// Vectorized: b128 global loads, bf16 hi/lo split, b64 LDS stores in
// WMMA-fragment order (k-block permute 0,2,1,3 per 8-wide block).
#define STAGE(q_, b_)                                                              \
    {                                                                              \
        const int g_  = (q_) >> 4;                                                 \
        const int kc_ = (q_) & 15;                                                 \
        const int kq  = (tid & 7) * 4;                                             \
        const int rr  = tid >> 3;                                                  \
        const int gg  = kq >> 3;                                                   \
        const int pb  = ((gg & 1) << 1) | (gg >> 1);                               \
        const int pk  = pb * 8 + (kq & 7);                                         \
        _Pragma("unroll")                                                          \
        for (int j = 0; j < 4; ++j) {                                              \
            int r = rr + j * 16;                                                   \
            f32x4 x = *(const f32x4*)&embeds[(size_t)(row0 + r) * C_DIM + kc_ * KC + kq]; \
            u16x4 h, l;                                                            \
            _Pragma("unroll")                                                      \
            for (int i = 0; i < 4; ++i) {                                          \
                unsigned short hh = f2bf(x[i]);                                    \
                h[i] = hh;                                                         \
                l[i] = f2bf(x[i] - bf2f(hh));                                      \
            }                                                                      \
            *(u16x4*)&Ahi[(b_) * ABUF + r * LDS_STR + pk] = h;                     \
            *(u16x4*)&Alo[(b_) * ABUF + r * LDS_STR + pk] = l;                     \
        }                                                                          \
        _Pragma("unroll")                                                          \
        for (int j = 0; j < 4; ++j) {                                              \
            int n = rr + j * 16;                                                   \
            f32x4 x = *(const f32x4*)&centroids[(size_t)(g_ * NG + n) * C_DIM + kc_ * KC + kq]; \
            u16x4 h, l;                                                            \
            _Pragma("unroll")                                                      \
            for (int i = 0; i < 4; ++i) {                                          \
                unsigned short hh = f2bf(x[i]);                                    \
                h[i] = hh;                                                         \
                l[i] = f2bf(x[i] - bf2f(hh));                                      \
            }                                                                      \
            *(u16x4*)&Bhi[(b_) * BBUF + n * LDS_STR + pk] = h;                     \
            *(u16x4*)&Blo[(b_) * BBUF + n * LDS_STR + pk] = l;                     \
        }                                                                          \
    }

__global__ __launch_bounds__(THREADS) void fused_dist_topk_kernel(
    const float* __restrict__ embeds,     // [M, C] row-major (M = B*N)
    const float* __restrict__ centroids,  // [P, C] row-major
    float* __restrict__ out)              // [M]
{
    __shared__ __align__(32) unsigned short Ahi[2 * ABUF];
    __shared__ __align__(32) unsigned short Alo[2 * ABUF];
    __shared__ __align__(32) unsigned short Bhi[2 * BBUF];
    __shared__ __align__(32) unsigned short Blo[2 * BBUF];
    __shared__ float centq[P_DIM];
    __shared__ float featq[M_BLK];
    __shared__ float mrg[WAVES][16][16][3];

    const int tid  = threadIdx.x;
    const int lane = tid & 31;
    const int wave = tid >> 5;
    const int hv   = lane >> 4;   // wave half (0: lanes 0-15, 1: lanes 16-31)
    const int l16  = lane & 15;
    const int row0 = blockIdx.x * M_BLK;

    // ---- feat[m] = sum_c embeds[m][c]^2  (exact f32, b128 coalesced) ----
    for (int r = 0; r < 16; ++r) {
        const float* ep = embeds + (size_t)(row0 + wave * 16 + r) * C_DIM;
        float s = 0.f;
        #pragma unroll
        for (int j = 0; j < 4; ++j) {
            f32x4 v = *(const f32x4*)&ep[lane * 4 + 128 * j];
            s += v[0] * v[0] + v[1] * v[1] + v[2] * v[2] + v[3] * v[3];
        }
        #pragma unroll
        for (int m = 16; m >= 1; m >>= 1) s += __shfl_xor(s, m, 32);
        if (lane == 0) featq[wave * 16 + r] = s;
    }
    // ---- cent[p] = sum_c centroids[p][c]^2 ----
    for (int p = wave; p < P_DIM; p += WAVES) {
        const float* cp = centroids + (size_t)p * C_DIM;
        float s = 0.f;
        #pragma unroll
        for (int j = 0; j < 4; ++j) {
            f32x4 v = *(const f32x4*)&cp[lane * 4 + 128 * j];
            s += v[0] * v[0] + v[1] * v[1] + v[2] * v[2] + v[3] * v[3];
        }
        #pragma unroll
        for (int m = 16; m >= 1; m >>= 1) s += __shfl_xor(s, m, 32);
        if (lane == 0) centq[p] = s;
    }

    // running top-3 of squared distance (sqrt deferred; monotone)
    float t0v[8], t1v[8], t2v[8];
    #pragma unroll
    for (int r = 0; r < 8; ++r) { t0v[r] = 1e30f; t1v[r] = 1e30f; t2v[r] = 1e30f; }

    v8f acc[NTILES];
    #pragma unroll
    for (int t = 0; t < NTILES; ++t)
        #pragma unroll
        for (int i = 0; i < 8; ++i) acc[t][i] = 0.f;

    // ---- software-pipelined main loop over 256 chunks (16 groups x 16 k) ----
    STAGE(0, 0);
    __syncthreads();  // also covers featq/centq

    const int arow = wave * 16 + l16;
    for (int q = 0; q < NCHUNKS; ++q) {
        const int cur = q & 1;
        if (q + 1 < NCHUNKS) STAGE(q + 1, cur ^ 1);   // overlaps with WMMA below

        v16bf ah = __builtin_bit_cast(v16bf,
            *(const v16u*)&Ahi[cur * ABUF + arow * LDS_STR + hv * 16]);
        v16bf al = __builtin_bit_cast(v16bf,
            *(const v16u*)&Alo[cur * ABUF + arow * LDS_STR + hv * 16]);
        #pragma unroll
        for (int t = 0; t < NTILES; ++t) {
            int n = t * 16 + l16;
            v16bf bh = __builtin_bit_cast(v16bf,
                *(const v16u*)&Bhi[cur * BBUF + n * LDS_STR + hv * 16]);
            v16bf bl = __builtin_bit_cast(v16bf,
                *(const v16u*)&Blo[cur * BBUF + n * LDS_STR + hv * 16]);
            acc[t] = __builtin_amdgcn_wmma_f32_16x16x32_bf16(false, ah, false, bh,
                                                             (short)0, acc[t], false, false);
            acc[t] = __builtin_amdgcn_wmma_f32_16x16x32_bf16(false, ah, false, bl,
                                                             (short)0, acc[t], false, false);
            acc[t] = __builtin_amdgcn_wmma_f32_16x16x32_bf16(false, al, false, bh,
                                                             (short)0, acc[t], false, false);
        }
        __syncthreads();

        if ((q & (KCHUNKS - 1)) == (KCHUNKS - 1)) {
            // ---- group epilogue: dist^2 + running top-3, reset accumulators ----
            const int g = q >> 4;
            float fq[8];
            #pragma unroll
            for (int r = 0; r < 8; ++r) fq[r] = featq[wave * 16 + hv * 8 + r];
            #pragma unroll
            for (int t = 0; t < NTILES; ++t) {
                float cq = centq[g * NG + t * 16 + l16];
                #pragma unroll
                for (int r = 0; r < 8; ++r) {
                    // C layout: VGPR r -> row (hv*8 + r), col l16 (+16*t)
                    float d2 = fq[r] + cq - 2.0f * acc[t][r];
                    TOP3_INSERT(d2, t0v[r], t1v[r], t2v[r]);
                    acc[t][r] = 0.f;
                }
            }
        }
    }

    // ---- cross-lane merge of top-3 candidates through LDS ----
    #pragma unroll
    for (int r = 0; r < 8; ++r) {
        int m = hv * 8 + r;
        mrg[wave][m][l16][0] = t0v[r];
        mrg[wave][m][l16][1] = t1v[r];
        mrg[wave][m][l16][2] = t2v[r];
    }
    __syncthreads();
    if (lane < 16) {
        float u0 = 1e30f, u1 = 1e30f, u2 = 1e30f;
        #pragma unroll
        for (int i = 0; i < 16; ++i) {
            #pragma unroll
            for (int k = 0; k < 3; ++k) {
                float d2 = mrg[wave][lane][i][k];
                TOP3_INSERT(d2, u0, u1, u2);
            }
        }
        float d0 = sqrtf(fmaxf(u0, 0.f));
        float d1 = sqrtf(fmaxf(u1, 0.f));
        float d2 = sqrtf(fmaxf(u2, 0.f));
        // softmin weight of nearest centroid times its distance
        float w0 = 1.0f / (1.0f + expf(d0 - d1) + expf(d0 - d2));
        out[row0 + wave * 16 + lane] = w0 * d0;
    }
}

extern "C" void kernel_launch(void* const* d_in, const int* in_sizes, int n_in,
                              void* d_out, int out_size, void* d_ws, size_t ws_size,
                              hipStream_t stream) {
    const float* embeds    = (const float*)d_in[0];   // [B*N, C] = [25088, 512]
    const float* centroids = (const float*)d_in[1];   // [P, C]   = [1024, 512]
    float* out = (float*)d_out;                       // [B*N] = [8,1,56,56] flat
    const int M = in_sizes[0] / C_DIM;                // 25088
    dim3 grid(M / M_BLK);                             // 392 workgroups
    fused_dist_topk_kernel<<<grid, THREADS, 0, stream>>>(embeds, centroids, out);
}
